// PhysicsInformedCSIModel_33852932227360
// MI455X (gfx1250) — matrix operand order repaired and checked by reading, not compiled, not run on hardware
//
#include <hip/hip_runtime.h>
#include <math.h>

typedef __attribute__((ext_vector_type(16))) _Float16 v16h;
typedef __attribute__((ext_vector_type(8)))  _Float16 v8h;
typedef __attribute__((ext_vector_type(8)))  float    v8f;

// ---------------------------------------------------------------------------
// WMMA helpers (CDNA5 f16 16x16x32, f32 accumulate)
// A operand: lane = M row (lane&15); elems 0..7 -> K = k0 + 8*(lane>>4) + j,
//            elems 8..15 -> K = k0 + 16 + 8*(lane>>4) + j   (ISA 7.12.2)
// B operand: lane = N (n0 + (lane&15)); elem j -> same K as A elem j
// C/D:       lane = N (lane&15); elem v -> M = 8*(lane>>4)+v
// ---------------------------------------------------------------------------
__device__ inline v8f wmma_f16(v16h a, v16h b, v8f c) {
  return __builtin_amdgcn_wmma_f32_16x16x32_f16(false, a, false, b, (short)0, c,
                                                false, false);
}

// A operand from f32 row, K multiple of 32 -> unconditional b128 loads.
__device__ inline v16h load_a_f32(const float* __restrict__ rowp, int k0,
                                  int lane) {
  const int kb = k0 + ((lane >> 4) << 3);
  float4 x0 = *(const float4*)(rowp + kb);
  float4 x1 = *(const float4*)(rowp + kb + 4);
  float4 y0 = *(const float4*)(rowp + kb + 16);
  float4 y1 = *(const float4*)(rowp + kb + 20);
  v16h r;
  r[0] = (_Float16)x0.x;  r[1] = (_Float16)x0.y;
  r[2] = (_Float16)x0.z;  r[3] = (_Float16)x0.w;
  r[4] = (_Float16)x1.x;  r[5] = (_Float16)x1.y;
  r[6] = (_Float16)x1.z;  r[7] = (_Float16)x1.w;
  r[8]  = (_Float16)y0.x; r[9]  = (_Float16)y0.y;
  r[10] = (_Float16)y0.z; r[11] = (_Float16)y0.w;
  r[12] = (_Float16)y1.x; r[13] = (_Float16)y1.y;
  r[14] = (_Float16)y1.z; r[15] = (_Float16)y1.w;
  return r;
}

// 16 contiguous f16 -> v16h (two b128 loads).
__device__ inline v16h load_contig16h(const _Float16* __restrict__ p) {
  v8h lo = *(const v8h*)p;
  v8h hi = *(const v8h*)(p + 8);
  return __builtin_shufflevector(lo, hi, 0, 1, 2, 3, 4, 5, 6, 7, 8, 9, 10, 11,
                                 12, 13, 14, 15);
}

// A operand from f16 LDS row (row stride padded so 16B alignment holds).
__device__ inline v16h load_a_lds(const _Float16* __restrict__ rowp, int k0,
                                  int lane) {
  const int kb = k0 + ((lane >> 4) << 3);
  v8h lo = *(const v8h*)(rowp + kb);
  v8h hi = *(const v8h*)(rowp + kb + 16);
  return __builtin_shufflevector(lo, hi, 0, 1, 2, 3, 4, 5, 6, 7, 8, 9, 10, 11,
                                 12, 13, 14, 15);
}

// B operand from pre-transposed f16 weights WT [K][N]: lane reads column
// n0 + (lane&15), elems in table K order (same as A).
__device__ inline v16h load_b_t(const _Float16* __restrict__ WT, int N, int n0,
                                int k0, int lane) {
  const _Float16* p = WT + n0 + (lane & 15);
  const int kb = k0 + ((lane >> 4) << 3);
  v16h r;
#pragma unroll
  for (int j = 0; j < 8; ++j) {
    r[j] = p[(size_t)(kb + j) * N];
    r[8 + j] = p[(size_t)(kb + 16 + j) * N];
  }
  return r;
}

__device__ inline float sigf(float x) { return 1.f / (1.f + __expf(-x)); }

// ---------------------------------------------------------------------------
// Weight transpose+convert: W [N,Kin] f32 -> WT [Kout,N] f16 (zero pad K).
// ---------------------------------------------------------------------------
__global__ void convert_wT(const float* __restrict__ W,
                           _Float16* __restrict__ WT, int N, int Kin,
                           int Kout) {
  int i = blockIdx.x * blockDim.x + threadIdx.x;
  if (i >= N * Kout) return;
  int n = i % N, k = i / N;
  WT[i] = (k < Kin) ? (_Float16)W[(size_t)n * Kin + k] : (_Float16)0.f;
}

// ---------------------------------------------------------------------------
// Pad x [64000,90] -> Xp [64000,96] (zeros in cols 90..95).
// ---------------------------------------------------------------------------
__global__ void pad_x(const float* __restrict__ x, float* __restrict__ xp) {
  size_t i = (size_t)blockIdx.x * blockDim.x + threadIdx.x;
  if (i >= (size_t)64000 * 96) return;
  int c = (int)(i % 96);
  size_t row = i / 96;
  xp[i] = (c < 90) ? x[row * 90 + c] : 0.f;
}

// ---------------------------------------------------------------------------
// Generic GEMM: Y[M,N] = act(X[M,K] @ W^T + b1[N] (+ b2[N])),
// weights pre-transposed as WT [K][N] f16.
// Requires K % 32 == 0 and N % 64 == 0.  act: 0 none, 1 relu, 2 elu+1
// Each wave computes a 16x64 block (1 M-tile x 4 N-tiles).
// ---------------------------------------------------------------------------
__global__ __launch_bounds__(256) void gemm_wmma(
    const float* __restrict__ X, const _Float16* __restrict__ WT,
    const float* __restrict__ b1, const float* __restrict__ b2,
    float* __restrict__ Y, int M, int N, int K, int act) {
  const int lane = threadIdx.x & 31;
  const int wave = threadIdx.x >> 5;
  const int ntN = N >> 6;                 // 64-wide N blocks
  const int ntiles = (M >> 4) * ntN;
  const int tile = blockIdx.x * 8 + wave;
  if (tile >= ntiles) return;
  const int m0 = (tile / ntN) << 4;
  const int n0 = (tile % ntN) << 6;
  const float* arow = X + (size_t)(m0 + (lane & 15)) * K;
  v8f acc0 = {0.f, 0.f, 0.f, 0.f, 0.f, 0.f, 0.f, 0.f};
  v8f acc1 = acc0, acc2 = acc0, acc3 = acc0;
  for (int k0 = 0; k0 < K; k0 += 32) {
    v16h a = load_a_f32(arow, k0, lane);
    v16h w0 = load_b_t(WT, N, n0, k0, lane);
    v16h w1 = load_b_t(WT, N, n0 + 16, k0, lane);
    v16h w2 = load_b_t(WT, N, n0 + 32, k0, lane);
    v16h w3 = load_b_t(WT, N, n0 + 48, k0, lane);
    acc0 = wmma_f16(a, w0, acc0);
    acc1 = wmma_f16(a, w1, acc1);
    acc2 = wmma_f16(a, w2, acc2);
    acc3 = wmma_f16(a, w3, acc3);
  }
  const int nh = lane & 15;
  const int mh = (lane >> 4) << 3;
  v8f accs[4] = {acc0, acc1, acc2, acc3};
#pragma unroll
  for (int q = 0; q < 4; ++q) {
    const int col = n0 + q * 16 + nh;
    float bias = b1[col] + (b2 ? b2[col] : 0.f);
#pragma unroll
    for (int v = 0; v < 8; ++v) {
      float y = accs[q][v] + bias;
      if (act == 1) y = fmaxf(y, 0.f);
      else if (act == 2) y = (y > 0.f) ? (y + 1.f) : __expf(y);
      Y[(size_t)(m0 + mh + v) * N + col] = y;
    }
  }
}

// ---------------------------------------------------------------------------
// LSTM recurrence: one workgroup per direction (blockIdx.x = dir).
// gx   : [2][B][Ts][512] precomputed x@Wih^T + bih + bhh
// WhhT : [2][128][512] f16 (pre-transposed)
// hout : [B][Ts][256]  (dir d writes columns d*128 .. d*128+127)
// Whh f16 tiles live in registers; h lives in LDS (f16); c in registers.
// ---------------------------------------------------------------------------
__global__ __launch_bounds__(1024) void lstm_scan(
    const float* __restrict__ gx, const _Float16* __restrict__ WhhT,
    float* __restrict__ hout, int Ts) {
  const int dir = blockIdx.x;
  const int tid = threadIdx.x;
  const int lane = tid & 31;
  const int wave = tid >> 5;
  const int mt = wave >> 3;  // 0..3  batch tile
  const int jt = wave & 7;   // 0..7  hidden tile
  const int b0 = mt << 4;
  const int j0 = jt << 4;

  __shared__ _Float16 hsh[64][136];  // 136 halves = 272B row stride (16B mult)

  const float* gxd = gx + (size_t)dir * 64 * (size_t)Ts * 512;
  const _Float16* WdT = WhhT + (size_t)dir * 128 * 512;

  // preload Whh tiles (B operand) into registers (strided f16 loads)
  v16h bw[4][4];
#pragma unroll
  for (int g = 0; g < 4; ++g)
#pragma unroll
    for (int ks = 0; ks < 4; ++ks)
      bw[g][ks] = load_b_t(WdT, 512, g * 128 + j0, ks * 32, lane);

  for (int i = tid; i < 64 * 136; i += 1024) (&hsh[0][0])[i] = (_Float16)0.f;

  v8f c;
#pragma unroll
  for (int v = 0; v < 8; ++v) c[v] = 0.f;

  const int nh = lane & 15;
  const int mh = (lane >> 4) << 3;
  const _Float16* arow = &hsh[b0 + (lane & 15)][0];

  __syncthreads();

  for (int ti = 0; ti < Ts; ++ti) {
    const int t = dir ? (Ts - 1 - ti) : ti;
    // read h_{t-1} (A operand) from LDS: 8 x ds_load_b128
    v16h a0 = load_a_lds(arow, 0, lane);
    v16h a1 = load_a_lds(arow, 32, lane);
    v16h a2 = load_a_lds(arow, 64, lane);
    v16h a3 = load_a_lds(arow, 96, lane);
    __syncthreads();  // everyone done reading h_{t-1}

    v8f acc[4];
#pragma unroll
    for (int g = 0; g < 4; ++g) {
#pragma unroll
      for (int v = 0; v < 8; ++v) {
        int b = b0 + mh + v;
        acc[g][v] = gxd[((size_t)b * Ts + t) * 512 + g * 128 + j0 + nh];
      }
      acc[g] = wmma_f16(a0, bw[g][0], acc[g]);
      acc[g] = wmma_f16(a1, bw[g][1], acc[g]);
      acc[g] = wmma_f16(a2, bw[g][2], acc[g]);
      acc[g] = wmma_f16(a3, bw[g][3], acc[g]);
    }

#pragma unroll
    for (int v = 0; v < 8; ++v) {
      float ig = sigf(acc[0][v]);
      float fg = sigf(acc[1][v]);
      float gg = tanhf(acc[2][v]);
      float og = sigf(acc[3][v]);
      float cn = fg * c[v] + ig * gg;
      c[v] = cn;
      float h = og * tanhf(cn);
      int b = b0 + mh + v;
      hsh[b][j0 + nh] = (_Float16)h;
      hout[((size_t)b * Ts + t) * 256 + dir * 128 + j0 + nh] = h;
    }
    __syncthreads();  // h_t visible for next step
  }
}

// ---------------------------------------------------------------------------
// avg-pool over time: out[b][tm][ch] = mean_{k<s} f[b][tm*s+k][ch]
// ---------------------------------------------------------------------------
__global__ void avgpool_k(const float* __restrict__ f, float* __restrict__ o,
                          int Tout, int s) {
  size_t i = (size_t)blockIdx.x * blockDim.x + threadIdx.x;
  size_t tot = (size_t)64 * Tout * 128;
  if (i >= tot) return;
  int ch = (int)(i % 128);
  size_t r = i / 128;
  int tm = (int)(r % Tout);
  int b = (int)(r / Tout);
  float acc = 0.f;
  for (int k = 0; k < s; ++k)
    acc += f[((size_t)b * 1000 + tm * s + k) * 128 + ch];
  o[i] = acc / (float)s;
}

__device__ inline float interp1(const float* __restrict__ hbuf, int Tin, int b,
                                int t, int ch) {
  float c = (t + 0.5f) * ((float)Tin / 1000.f) - 0.5f;
  c = fminf(fmaxf(c, 0.f), (float)(Tin - 1));
  int lo = (int)floorf(c);
  int hi = min(lo + 1, Tin - 1);
  float w = c - (float)lo;
  const float* base = hbuf + ((size_t)b * Tin) * 256 + ch;
  return base[(size_t)lo * 256] * (1.f - w) + base[(size_t)hi * 256] * w;
}

__global__ void fuse_interp(const float* __restrict__ hf,
                            const float* __restrict__ hm,
                            const float* __restrict__ hc,
                            const float* __restrict__ w3,
                            float* __restrict__ out) {
  size_t i = (size_t)blockIdx.x * blockDim.x + threadIdx.x;
  if (i >= (size_t)64 * 1000 * 256) return;
  int ch = (int)(i % 256);
  size_t r = i / 256;
  int t = (int)(r % 1000);
  int b = (int)(r / 1000);
  float w0 = w3[0], w1 = w3[1], w2 = w3[2];
  float mx = fmaxf(w0, fmaxf(w1, w2));
  float e0 = __expf(w0 - mx), e1 = __expf(w1 - mx), e2 = __expf(w2 - mx);
  float inv = 1.f / (e0 + e1 + e2);
  float vf = hf[((size_t)b * 1000 + t) * 256 + ch];
  float vm = interp1(hm, 200, b, t, ch);
  float vc = interp1(hc, 100, b, t, ch);
  out[i] = (e0 * vf + e1 * vm + e2 * vc) * inv;
}

// ---------------------------------------------------------------------------
// Performer-style linear attention per (b, head). HD=16, T=1000.
// ---------------------------------------------------------------------------
__global__ __launch_bounds__(256) void linattn(const float* __restrict__ Q,
                                               const float* __restrict__ K,
                                               const float* __restrict__ V,
                                               float* __restrict__ O) {
  const int b = blockIdx.x, h = blockIdx.y;
  __shared__ float kv[16][17];
  __shared__ float ksum[16];
  __shared__ float kt[64][16];
  __shared__ float vt[64][16];
  const int tid = threadIdx.x;
  const int d = tid >> 4, e = tid & 15;
  float acc = 0.f, ks = 0.f;
  for (int t0 = 0; t0 < 1000; t0 += 64) {
    int cn = min(64, 1000 - t0);
    for (int i = tid; i < cn * 16; i += 256) {
      int tt = i >> 4, cc = i & 15;
      size_t rowoff = ((size_t)(b * 1000 + t0 + tt)) * 128 + h * 16 + cc;
      kt[tt][cc] = K[rowoff];
      vt[tt][cc] = V[rowoff];
    }
    __syncthreads();
    for (int tt = 0; tt < cn; ++tt) {
      float kd = kt[tt][d];
      acc += kd * vt[tt][e];
      if (e == 0) ks += kd;
    }
    __syncthreads();
  }
  kv[d][e] = acc;
  if (e == 0) ksum[d] = ks;
  __syncthreads();
  for (int i = tid; i < 1000 * 16; i += 256) {
    int t = i >> 4, ee = i & 15;
    const float* q = Q + ((size_t)(b * 1000 + t)) * 128 + h * 16;
    float num = 0.f, den = 0.f;
#pragma unroll
    for (int dd = 0; dd < 16; ++dd) {
      float qd = q[dd];
      num += qd * kv[dd][ee];
      den += qd * ksum[dd];
    }
    O[((size_t)(b * 1000 + t)) * 128 + h * 16 + ee] = num / (den + 1e-6f);
  }
}

__global__ void mean_t(const float* __restrict__ a, float* __restrict__ p) {
  int i = blockIdx.x * blockDim.x + threadIdx.x;
  if (i >= 64 * 128) return;
  int ch = i & 127, b = i >> 7;
  float s = 0.f;
  for (int t = 0; t < 1000; ++t) s += a[((size_t)b * 1000 + t) * 128 + ch];
  p[i] = s * (1.f / 1000.f);
}

__global__ void cls2(const float* __restrict__ hc, const float* __restrict__ W,
                     const float* __restrict__ bias, float* __restrict__ out) {
  int i = blockIdx.x * blockDim.x + threadIdx.x;
  if (i >= 64 * 6) return;
  int n = i % 6, b = i / 6;
  float s = bias[n];
  for (int k = 0; k < 64; ++k) s += hc[b * 64 + k] * W[n * 64 + k];
  out[i] = s;
}

// ---------------------------------------------------------------------------
// Host side
// ---------------------------------------------------------------------------
static void gemm_launch(const float* X, const _Float16* WT, const float* b1,
                        const float* b2, float* Y, int M, int N, int K, int act,
                        hipStream_t s) {
  int ntiles = (M / 16) * (N / 64);
  int blocks = (ntiles + 7) / 8;
  gemm_wmma<<<blocks, 256, 0, s>>>(X, WT, b1, b2, Y, M, N, K, act);
}

static void wconv(const float* W, _Float16* WT, int N, int Kin, int Kout,
                  hipStream_t s) {
  int tot = N * Kout;
  convert_wT<<<(tot + 255) / 256, 256, 0, s>>>(W, WT, N, Kin, Kout);
}

static void bilstm(void* const* d_in, int pb, const _Float16* wih0t,
                   const _Float16* whh0t, const _Float16* wih1t,
                   const _Float16* whh1t, const float* Xin, int Ts,
                   float* hcat0, float* hout, float* gx, hipStream_t s) {
  const float* bih0 = (const float*)d_in[pb + 2];
  const float* bhh0 = (const float*)d_in[pb + 3];
  const float* bih1 = (const float*)d_in[pb + 6];
  const float* bhh1 = (const float*)d_in[pb + 7];
  size_t gstride = (size_t)64 * Ts * 512;
  for (int d = 0; d < 2; ++d)
    gemm_launch(Xin, wih0t + (size_t)d * 128 * 512, bih0 + d * 512,
                bhh0 + d * 512, gx + d * gstride, 64 * Ts, 512, 128, 0, s);
  lstm_scan<<<2, 1024, 0, s>>>(gx, whh0t, hcat0, Ts);
  for (int d = 0; d < 2; ++d)
    gemm_launch(hcat0, wih1t + (size_t)d * 256 * 512, bih1 + d * 512,
                bhh1 + d * 512, gx + d * gstride, 64 * Ts, 512, 256, 0, s);
  lstm_scan<<<2, 1024, 0, s>>>(gx, whh1t, hout, Ts);
}

extern "C" void kernel_launch(void* const* d_in, const int* in_sizes, int n_in,
                              void* d_out, int out_size, void* d_ws,
                              size_t ws_size, hipStream_t stream) {
  (void)in_sizes; (void)n_in; (void)out_size; (void)ws_size;
  const float* x = (const float*)d_in[0];
  float* ws = (float*)d_ws;

  // workspace layout (floats)
  float* F  = ws;                 //  8,192,000  projected input [B,1000,128]
  float* XM = F + 8192000;        //  1,638,400  pooled med
  float* XC = XM + 1638400;       //    819,200  pooled coarse
  float* H0 = XC + 819200;        // 16,384,000  layer-0 concat / fused
  float* HF = H0 + 16384000;      // 16,384,000  h_fine
  float* HM = HF + 16384000;      //  3,276,800  h_med
  float* HC = HM + 3276800;       //  1,638,400  h_coarse
  float* GX = HC + 1638400;       // 65,536,000  gate precompute / post buffers
  float* PO = GX + 65536000;      //      8,192  pooled
  float* HL = PO + 8192;          //      4,096  hcls
  // pre-LSTM temporaries alias the (then unused) GX region
  float* F1 = GX;                 // 16,384,000  relu(x@W1^T+b1)
  float* XP = GX + 16384000;      //  6,144,000  padded input [64000,96]
  // post-LSTM buffers alias the (then dead) GX region
  float* LO = GX;
  float* QB = LO + 8192000;
  float* KB = QB + 8192000;
  float* VB = KB + 8192000;
  float* AO = VB + 8192000;
  float* AT = AO + 8192000;

  // pre-transposed f16 weights live after HL (~4.3 MB total)
  _Float16* wt = (_Float16*)(HL + 4096);
  _Float16* W1T = wt;            wt += 96 * 256;
  _Float16* W2T = wt;            wt += 256 * 128;
  _Float16* LSTMT[3][4];
  for (int s = 0; s < 3; ++s) {
    LSTMT[s][0] = wt; wt += 2 * 128 * 512;  // Wih0T
    LSTMT[s][1] = wt; wt += 2 * 128 * 512;  // Whh0T
    LSTMT[s][2] = wt; wt += 2 * 256 * 512;  // Wih1T
    LSTMT[s][3] = wt; wt += 2 * 128 * 512;  // Whh1T
  }
  _Float16* OWT = wt;            wt += 256 * 128;
  _Float16* QWT = wt;            wt += 128 * 128;
  _Float16* KWT = wt;            wt += 128 * 128;
  _Float16* VWT = wt;            wt += 128 * 128;
  _Float16* AWT = wt;            wt += 128 * 128;
  _Float16* C1T = wt;            wt += 128 * 64;

  // 0a) transpose/convert all weights to [K][N] f16
  wconv((const float*)d_in[1], W1T, 256, 90, 96, stream);
  wconv((const float*)d_in[3], W2T, 128, 256, 256, stream);
  for (int s = 0; s < 3; ++s) {
    int pb = 5 + 8 * s;
    for (int d = 0; d < 2; ++d) {
      wconv((const float*)d_in[pb + 0] + (size_t)d * 512 * 128,
            LSTMT[s][0] + (size_t)d * 128 * 512, 512, 128, 128, stream);
      wconv((const float*)d_in[pb + 1] + (size_t)d * 512 * 128,
            LSTMT[s][1] + (size_t)d * 128 * 512, 512, 128, 128, stream);
      wconv((const float*)d_in[pb + 4] + (size_t)d * 512 * 256,
            LSTMT[s][2] + (size_t)d * 256 * 512, 512, 256, 256, stream);
      wconv((const float*)d_in[pb + 5] + (size_t)d * 512 * 128,
            LSTMT[s][3] + (size_t)d * 128 * 512, 512, 128, 128, stream);
    }
  }
  wconv((const float*)d_in[30], OWT, 128, 256, 256, stream);
  wconv((const float*)d_in[32], QWT, 128, 128, 128, stream);
  wconv((const float*)d_in[34], KWT, 128, 128, 128, stream);
  wconv((const float*)d_in[36], VWT, 128, 128, 128, stream);
  wconv((const float*)d_in[38], AWT, 128, 128, 128, stream);
  wconv((const float*)d_in[40], C1T, 64, 128, 128, stream);

  // 0b) pad x to K=96
  pad_x<<<24000, 256, 0, stream>>>(x, XP);
  // 1) input projection
  gemm_launch(XP, W1T, (const float*)d_in[2], nullptr, F1, 64000, 256, 96, 1,
              stream);
  gemm_launch(F1, W2T, (const float*)d_in[4], nullptr, F, 64000, 128, 256, 0,
              stream);
  // 2) multi-scale pooling
  avgpool_k<<<6400, 256, 0, stream>>>(F, XM, 200, 5);
  avgpool_k<<<3200, 256, 0, stream>>>(F, XC, 100, 10);
  // 3) bidirectional LSTMs
  bilstm(d_in, 5, LSTMT[0][0], LSTMT[0][1], LSTMT[0][2], LSTMT[0][3], F, 1000,
         H0, HF, GX, stream);
  bilstm(d_in, 13, LSTMT[1][0], LSTMT[1][1], LSTMT[1][2], LSTMT[1][3], XM, 200,
         H0, HM, GX, stream);
  bilstm(d_in, 21, LSTMT[2][0], LSTMT[2][1], LSTMT[2][2], LSTMT[2][3], XC, 100,
         H0, HC, GX, stream);
  // 4) interpolate + softmax-weighted fusion (into H0)
  fuse_interp<<<64000, 256, 0, stream>>>(HF, HM, HC, (const float*)d_in[29],
                                         H0);
  // 5) output projection
  gemm_launch(H0, OWT, (const float*)d_in[31], nullptr, LO, 64000, 128, 256, 0,
              stream);
  // 6) q,k,v (elu+1 feature map on q,k)
  gemm_launch(LO, QWT, (const float*)d_in[33], nullptr, QB, 64000, 128, 128, 2,
              stream);
  gemm_launch(LO, KWT, (const float*)d_in[35], nullptr, KB, 64000, 128, 128, 2,
              stream);
  gemm_launch(LO, VWT, (const float*)d_in[37], nullptr, VB, 64000, 128, 128, 0,
              stream);
  // 7) linear attention
  linattn<<<dim3(64, 8), 256, 0, stream>>>(QB, KB, VB, AO);
  // 8) attention output projection
  gemm_launch(AO, AWT, (const float*)d_in[39], nullptr, AT, 64000, 128, 128, 0,
              stream);
  // 9) temporal mean + classifier
  mean_t<<<32, 256, 0, stream>>>(AT, PO);
  gemm_launch(PO, C1T, (const float*)d_in[41], nullptr, HL, 64, 64, 128, 1,
              stream);
  cls2<<<2, 256, 0, stream>>>(HL, (const float*)d_in[42],
                              (const float*)d_in[43], (float*)d_out);
}
